// Attention_29592324670129
// MI455X (gfx1250) — matrix operand order repaired and checked by reading, not compiled, not run on hardware
//
#include <hip/hip_runtime.h>
#include <hip/hip_bf16.h>
#include <stdint.h>

// Problem constants (match reference)
#define BATCH 2
#define SEQ   2048
#define CDIM  1024
#define NHEAD 16
#define HDIM  64
#define QKV3  3072   // 3*CDIM

typedef __attribute__((ext_vector_type(16))) __bf16 bf16x16;
typedef __attribute__((ext_vector_type(8)))  __bf16 bf16x8;
typedef __attribute__((ext_vector_type(8)))  float  f32x8;

static __device__ __forceinline__ bf16x16 concat8(bf16x8 a, bf16x8 b) {
  return __builtin_shufflevector(a, b, 0,1,2,3,4,5,6,7,8,9,10,11,12,13,14,15);
}

static __device__ __forceinline__ f32x8 wmma_bf16(bf16x16 a, bf16x16 b, f32x8 c) {
  // D = A(16x32) * B(32x16) + C, f32 accumulate
  return __builtin_amdgcn_wmma_f32_16x16x32_bf16(false, a, false, b, (short)0, c,
                                                 false, false);
}

// gfx1250 async copy: 16 bytes global -> LDS per lane, tracked by ASYNCcnt.
// lds_off = workgroup-relative LDS byte address (low 32 bits of flat shared ptr).
static __device__ __forceinline__ void async_copy_b128(uint32_t lds_off, const void* g) {
  asm volatile("global_load_async_to_lds_b128 %0, %1, off"
               :: "v"(lds_off), "v"(g) : "memory");
}
static __device__ __forceinline__ void wait_asynccnt0() {
  asm volatile("s_wait_asynccnt 0x0" ::: "memory");
}

// ---------------------------------------------------------------------------
// Kernel 0a: f32 -> bf16 convert (x)
// ---------------------------------------------------------------------------
__global__ __launch_bounds__(256) void cvt_f32_bf16(const float* __restrict__ in,
                                                    __bf16* __restrict__ out, int n) {
  int i = blockIdx.x * 256 + threadIdx.x;
  if (i < n) out[i] = (__bf16)in[i];
}

// ---------------------------------------------------------------------------
// Kernel 0b: transpose + convert weights.  in: (R rows, Cc cols) f32 row-major
// out: (Cc rows, R cols) bf16 row-major  (out[c*R + r] = in[r*Cc + c])
// ---------------------------------------------------------------------------
__global__ __launch_bounds__(256) void transpose_bf16(const float* __restrict__ in,
                                                      __bf16* __restrict__ out,
                                                      int R, int Cc) {
  __shared__ float tile[16][17];
  int c0 = blockIdx.x * 16, r0 = blockIdx.y * 16;
  int tx = threadIdx.x & 15, ty = threadIdx.x >> 4;  // 256 thr -> 16x16
  tile[ty][tx] = in[(size_t)(r0 + ty) * Cc + (c0 + tx)];
  __syncthreads();
  out[(size_t)(c0 + ty) * R + (r0 + tx)] = (__bf16)tile[tx][ty];
}

// ---------------------------------------------------------------------------
// Kernel 1: QKV GEMM, block-cooperative.  Block = 8 waves sharing one 64-col
// B strip; B tile (64x32 bf16 = 4KB) double-buffered in LDS via async copies;
// A fragments software-pipelined one k-step ahead in registers.
// Each wave computes a 16(M) x 64(N) strip.  Q,K as (B,H,N,D); V as (B,H,D,N).
// ---------------------------------------------------------------------------
__global__ __launch_bounds__(256) void qkv_gemm(const __bf16* __restrict__ xb,
                                                const __bf16* __restrict__ wT,
                                                __bf16* __restrict__ qb,
                                                __bf16* __restrict__ kbuf,
                                                __bf16* __restrict__ vtb) {
  __shared__ __bf16 tileB[2][64][32];             // 2 x 4KB double buffer
  const int NT = QKV3 / 64;                       // 48 N-strips
  int tid = threadIdx.x;
  int w = tid >> 5, lane = tid & 31;
  int half = lane >> 4, l16 = lane & 15;
  int nt = blockIdx.x % NT;
  int mt = (blockIdx.x / NT) * 8 + w;             // 8 M-tiles per block

  // async-copy assignment: thread -> (col 0..63, 8-elem chunk 0..3)
  int cc = tid >> 2;
  int kk = (tid & 3) * 8;
  const __bf16* gB = wT + (size_t)(nt * 64 + cc) * CDIM + kk;
  uint32_t myLds = (uint32_t)(uintptr_t)&tileB[0][0][0] + (uint32_t)((cc * 32 + kk) * 2);
  const uint32_t bufStride = 64 * 32 * 2;         // 4096 B

  async_copy_b128(myLds, gB);                     // k-step 0 -> buf 0
  wait_asynccnt0();
  __syncthreads();

  f32x8 acc[4] = {};
  const __bf16* aRow = xb + (size_t)(mt * 16 + l16) * CDIM;
  // A fragment pipeline (one k-step ahead)
  bf16x8 a0 = *(const bf16x8*)(aRow + half * 8);
  bf16x8 a1 = *(const bf16x8*)(aRow + 16 + half * 8);
  const int steps = CDIM / 32;
  for (int i = 0; i < steps; ++i) {
    int cur = i & 1;
    bf16x8 a0n, a1n;
    if (i + 1 < steps) {                          // prefetch next B tile + A frag
      async_copy_b128(myLds + (uint32_t)(cur ^ 1) * bufStride, gB + (i + 1) * 32);
      a0n = *(const bf16x8*)(aRow + (i + 1) * 32 + half * 8);
      a1n = *(const bf16x8*)(aRow + (i + 1) * 32 + 16 + half * 8);
    }
    bf16x16 a = concat8(a0, a1);
    bf16x16 bfr[4];
#pragma unroll
    for (int j = 0; j < 4; ++j) {                 // batch all B-frag LDS loads
      const __bf16* bRow = &tileB[cur][j * 16 + l16][half * 8];
      bfr[j] = concat8(*(const bf16x8*)bRow, *(const bf16x8*)(bRow + 16));
    }
#pragma unroll
    for (int j = 0; j < 4; ++j)                   // back-to-back WMMAs
      acc[j] = wmma_bf16(a, bfr[j], acc[j]);
    if (i + 1 < steps) {
      wait_asynccnt0();
      __syncthreads();
      a0 = a0n; a1 = a1n;
    }
  }

  // scatter (col strip of 64 stays inside one {q,k,v} segment and one head)
#pragma unroll
  for (int j = 0; j < 4; ++j) {
    int col = nt * 64 + j * 16 + l16;
    int s = col / CDIM, rem = col % CDIM;
    int h = rem / HDIM, d = rem % HDIM;
#pragma unroll
    for (int vi = 0; vi < 8; ++vi) {
      int r = mt * 16 + vi + 8 * half;
      int b = r / SEQ, n = r % SEQ;
      __bf16 val = (__bf16)acc[j][vi];
      size_t bh = (size_t)(b * NHEAD + h);
      if (s == 0)      qb[(bh * SEQ + n) * HDIM + d]   = val;
      else if (s == 1) kbuf[(bh * SEQ + n) * HDIM + d] = val;
      else             vtb[(bh * HDIM + d) * SEQ + n]  = val;
    }
  }
}

// ---------------------------------------------------------------------------
// Kernel 2: flash attention, block-cooperative.  All 8 waves of a block share
// one (b,h): K tile (32x64) and V tile (64x32) for each 32-key block are
// staged in LDS via double-buffered async copies and shared block-wide.
// Each wave owns one 16-query tile with online softmax.
// ---------------------------------------------------------------------------
__global__ __launch_bounds__(256) void flash_attn(const __bf16* __restrict__ qb,
                                                  const __bf16* __restrict__ kbuf,
                                                  const __bf16* __restrict__ vtb,
                                                  const float* __restrict__ f,
                                                  float* __restrict__ attn_out,
                                                  __bf16* __restrict__ fusedb) {
  __shared__ __bf16 tileK[2][32][64];             // 2 x 4KB
  __shared__ __bf16 tileV[2][64][32];             // 2 x 4KB
  __shared__ __bf16 ldsP[8][16][32];              // per-wave P staging, 8KB
  int tid = threadIdx.x;
  int lane = tid & 31;
  int w = tid >> 5;
  int wave = blockIdx.x * 8 + w;
  int qt = wave & 127;                            // SEQ/16 = 128 query tiles
  int bh = wave >> 7;                             // same for all 8 waves/block
  int half = lane >> 4, l16 = lane & 15;
  int q0 = qt * 16;

  const __bf16* qbase = qb   + (size_t)bh * SEQ * HDIM;
  const __bf16* kbase = kbuf + (size_t)bh * SEQ * HDIM;
  const __bf16* vbase = vtb  + (size_t)bh * HDIM * SEQ;

  // async-copy assignments (per thread, 16B each for K and V)
  int kRow = tid >> 3, kChunk = (tid & 7) * 8;    // K: row 0..31, d-chunk
  int vRow = tid >> 2, vChunk = (tid & 3) * 8;    // V: d-row 0..63, key-chunk
  const __bf16* gK = kbase + (size_t)kRow * HDIM + kChunk;
  const __bf16* gV = vbase + (size_t)vRow * SEQ + vChunk;
  uint32_t ldsK = (uint32_t)(uintptr_t)&tileK[0][0][0] + (uint32_t)((kRow * 64 + kChunk) * 2);
  uint32_t ldsV = (uint32_t)(uintptr_t)&tileV[0][0][0] + (uint32_t)((vRow * 32 + vChunk) * 2);
  const uint32_t bufK = 32 * 64 * 2, bufV = 64 * 32 * 2;   // 4096 B each

  // Q A-fragments for both 32-wide K-steps of D=64, kept in registers
  bf16x16 qa[2];
  {
    const __bf16* qrow = qbase + (size_t)(q0 + l16) * HDIM;
#pragma unroll
    for (int s = 0; s < 2; ++s)
      qa[s] = concat8(*(const bf16x8*)(qrow + s * 32 + half * 8),
                      *(const bf16x8*)(qrow + s * 32 + 16 + half * 8));
  }

  async_copy_b128(ldsK, gK);                      // key block 0 -> buf 0
  async_copy_b128(ldsV, gV);
  wait_asynccnt0();
  __syncthreads();

  f32x8 acc[4] = {};
  float mrow[8], lrow[8];
#pragma unroll
  for (int vi = 0; vi < 8; ++vi) { mrow[vi] = -1e30f; lrow[vi] = 0.f; }
  const float scale = 0.125f;                     // D^-0.5 = 1/8

  const int steps = SEQ / 32;                     // 64 key blocks
  for (int i = 0; i < steps; ++i) {
    int cur = i & 1;
    if (i + 1 < steps) {                          // prefetch next K/V tiles
      async_copy_b128(ldsK + (uint32_t)(cur ^ 1) * bufK, gK + (size_t)(i + 1) * 32 * HDIM);
      async_copy_b128(ldsV + (uint32_t)(cur ^ 1) * bufV, gV + (i + 1) * 32);
    }
    // ---- scores: S(16x32) = Q(16x64) . K^T (K from LDS) -----------------
    bf16x16 kf[4];
#pragma unroll
    for (int ntile = 0; ntile < 2; ++ntile)
#pragma unroll
      for (int s = 0; s < 2; ++s) {
        const __bf16* kr = &tileK[cur][ntile * 16 + l16][s * 32 + half * 8];
        kf[ntile * 2 + s] = concat8(*(const bf16x8*)kr, *(const bf16x8*)(kr + 16));
      }
    f32x8 sc[2] = {};
#pragma unroll
    for (int ntile = 0; ntile < 2; ++ntile)
#pragma unroll
      for (int s = 0; s < 2; ++s)
        sc[ntile] = wmma_bf16(qa[s], kf[ntile * 2 + s], sc[ntile]);
    // ---- online softmax (row = vi + 8*half, cols striped over 16 lanes) -
#pragma unroll
    for (int vi = 0; vi < 8; ++vi) {
      float s0 = sc[0][vi] * scale, s1 = sc[1][vi] * scale;
      float bm = fmaxf(s0, s1);
#pragma unroll
      for (int m = 1; m < 16; m <<= 1) bm = fmaxf(bm, __shfl_xor(bm, m, 32));
      float newm = fmaxf(mrow[vi], bm);
      float corr = __expf(mrow[vi] - newm);
      float p0 = __expf(s0 - newm), p1 = __expf(s1 - newm);
      float rs = p0 + p1;
#pragma unroll
      for (int m = 1; m < 16; m <<= 1) rs += __shfl_xor(rs, m, 32);
      lrow[vi] = lrow[vi] * corr + rs;
      mrow[vi] = newm;
#pragma unroll
      for (int dt = 0; dt < 4; ++dt) acc[dt][vi] *= corr;
      int mr = vi + 8 * half;
      ldsP[w][mr][l16]      = (__bf16)p0;
      ldsP[w][mr][16 + l16] = (__bf16)p1;
    }
    asm volatile("s_wait_dscnt 0x0" ::: "memory");  // P stores -> P loads
    bf16x16 pa = concat8(*(const bf16x8*)&ldsP[w][l16][half * 8],
                         *(const bf16x8*)&ldsP[w][l16][16 + half * 8]);
    // ---- O += P(16x32) . V(32x64) (V from LDS) --------------------------
    bf16x16 vf[4];
#pragma unroll
    for (int dt = 0; dt < 4; ++dt) {
      const __bf16* vr = &tileV[cur][dt * 16 + l16][half * 8];
      vf[dt] = concat8(*(const bf16x8*)vr, *(const bf16x8*)(vr + 16));
    }
#pragma unroll
    for (int dt = 0; dt < 4; ++dt)
      acc[dt] = wmma_bf16(pa, vf[dt], acc[dt]);
    if (i + 1 < steps) {
      wait_asynccnt0();
      __syncthreads();
    }
  }

  int b = bh >> 4, h = bh & 15;
#pragma unroll
  for (int dt = 0; dt < 4; ++dt) {
#pragma unroll
    for (int vi = 0; vi < 8; ++vi) {
      int row = q0 + vi + 8 * half;
      int col = h * HDIM + dt * 16 + l16;
      size_t off = (size_t)(b * SEQ + row) * CDIM + col;
      float val = acc[dt][vi] / lrow[vi];
      attn_out[off] = val;                            // attn2gcn (output 2)
      fusedb[off] = (__bf16)(val + f[off]);           // residual fuse -> bf16
    }
  }
}

// ---------------------------------------------------------------------------
// Kernel 3: output projection, block-cooperative like kernel 1.
// fused(4096,1024)bf16 @ WprojT(1024,1024)bf16 + bias -> out f32.
// ---------------------------------------------------------------------------
__global__ __launch_bounds__(256) void proj_gemm(const __bf16* __restrict__ fusedb,
                                                 const __bf16* __restrict__ wT,
                                                 const float* __restrict__ bias,
                                                 float* __restrict__ out) {
  __shared__ __bf16 tileB[2][64][32];
  const int NT = CDIM / 64;                       // 16 N-strips
  int tid = threadIdx.x;
  int w = tid >> 5, lane = tid & 31;
  int half = lane >> 4, l16 = lane & 15;
  int nt = blockIdx.x % NT;
  int mt = (blockIdx.x / NT) * 8 + w;

  int cc = tid >> 2;
  int kk = (tid & 3) * 8;
  const __bf16* gB = wT + (size_t)(nt * 64 + cc) * CDIM + kk;
  uint32_t myLds = (uint32_t)(uintptr_t)&tileB[0][0][0] + (uint32_t)((cc * 32 + kk) * 2);
  const uint32_t bufStride = 64 * 32 * 2;

  async_copy_b128(myLds, gB);
  wait_asynccnt0();
  __syncthreads();

  f32x8 acc[4] = {};
  const __bf16* aRow = fusedb + (size_t)(mt * 16 + l16) * CDIM;
  bf16x8 a0 = *(const bf16x8*)(aRow + half * 8);
  bf16x8 a1 = *(const bf16x8*)(aRow + 16 + half * 8);
  const int steps = CDIM / 32;
  for (int i = 0; i < steps; ++i) {
    int cur = i & 1;
    bf16x8 a0n, a1n;
    if (i + 1 < steps) {
      async_copy_b128(myLds + (uint32_t)(cur ^ 1) * bufStride, gB + (i + 1) * 32);
      a0n = *(const bf16x8*)(aRow + (i + 1) * 32 + half * 8);
      a1n = *(const bf16x8*)(aRow + (i + 1) * 32 + 16 + half * 8);
    }
    bf16x16 a = concat8(a0, a1);
    bf16x16 bfr[4];
#pragma unroll
    for (int j = 0; j < 4; ++j) {
      const __bf16* bRow = &tileB[cur][j * 16 + l16][half * 8];
      bfr[j] = concat8(*(const bf16x8*)bRow, *(const bf16x8*)(bRow + 16));
    }
#pragma unroll
    for (int j = 0; j < 4; ++j)
      acc[j] = wmma_bf16(a, bfr[j], acc[j]);
    if (i + 1 < steps) {
      wait_asynccnt0();
      __syncthreads();
      a0 = a0n; a1 = a1n;
    }
  }
#pragma unroll
  for (int j = 0; j < 4; ++j) {
    int col = nt * 64 + j * 16 + l16;
    float bs = bias[col];
#pragma unroll
    for (int vi = 0; vi < 8; ++vi) {
      int r = mt * 16 + vi + 8 * half;
      out[(size_t)r * CDIM + col] = acc[j][vi] + bs;
    }
  }
}

// ---------------------------------------------------------------------------
extern "C" void kernel_launch(void* const* d_in, const int* in_sizes, int n_in,
                              void* d_out, int out_size, void* d_ws, size_t ws_size,
                              hipStream_t stream) {
  const float* x      = (const float*)d_in[0];   // (B,N,C)
  const float* f      = (const float*)d_in[1];   // (B,N,C)
  const float* W_qkv  = (const float*)d_in[2];   // (C,3C)
  const float* W_proj = (const float*)d_in[3];   // (C,C)
  const float* b_proj = (const float*)d_in[4];   // (C,)

  float* out_proj = (float*)d_out;                                 // (B,N,C)
  float* attn_out = (float*)d_out + (size_t)BATCH * SEQ * CDIM;    // (B,N,C)

  // Workspace layout (bf16 buffers), 48 MiB total
  char* ws = (char*)d_ws;
  __bf16* xb     = (__bf16*)(ws);                        // 8 MiB  x bf16
  __bf16* wqkvT  = (__bf16*)(ws + (size_t)8  * 1048576); // 6 MiB
  __bf16* wprojT = (__bf16*)(ws + (size_t)14 * 1048576); // 2 MiB
  __bf16* qb     = (__bf16*)(ws + (size_t)16 * 1048576); // 8 MiB (B,H,N,D)
  __bf16* kbuf   = (__bf16*)(ws + (size_t)24 * 1048576); // 8 MiB (B,H,N,D)
  __bf16* vtb    = (__bf16*)(ws + (size_t)32 * 1048576); // 8 MiB (B,H,D,N)
  __bf16* fusedb = (__bf16*)(ws + (size_t)40 * 1048576); // 8 MiB (B,N,C)

  // 0) precision / layout prep
  int nx = BATCH * SEQ * CDIM;
  cvt_f32_bf16<<<nx / 256, 256, 0, stream>>>(x, xb, nx);
  transpose_bf16<<<dim3(QKV3 / 16, CDIM / 16), 256, 0, stream>>>(W_qkv, wqkvT, CDIM, QKV3);
  transpose_bf16<<<dim3(CDIM / 16, CDIM / 16), 256, 0, stream>>>(W_proj, wprojT, CDIM, CDIM);

  // 1) QKV projection: 32 M-groups x 48 N-strips = 1536 blocks
  qkv_gemm<<<32 * 48, 256, 0, stream>>>(xb, wqkvT, qb, kbuf, vtb);

  // 2) flash attention: 32 (b,h) x 16 qt-groups = 512 blocks (8 waves each)
  flash_attn<<<4096 / 8, 256, 0, stream>>>(qb, kbuf, vtb, f, attn_out, fusedb);

  // 3) output projection: 32 M-groups x 16 N-strips = 512 blocks
  proj_gemm<<<32 * 16, 256, 0, stream>>>(fusedb, wprojT, b_proj, out_proj);
}
